// Mamba2Metal_12300786336335
// MI455X (gfx1250) — matrix operand order repaired and checked by reading, not compiled, not run on hardware
//
#include <hip/hip_runtime.h>
#include <hip/hip_bf16.h>

// ---------------- problem constants (match reference) ----------------
#define D_MODEL   1024
#define D_INNER   2048
#define HEADDIM   64
#define NHEADS    32
#define D_STATE   128
#define D_CONV    4
#define CHUNK     256
#define CONV_DIM  (D_INNER + 2*D_STATE)             // 2304
#define D_IN_PROJ (2*D_INNER + 2*D_STATE + NHEADS)  // 4384
#define NPROJ_PAD 4480                              // 4384 padded up to /128
#define BATCH     2
#define SEQLEN    4096
#define NCHUNK    (SEQLEN/CHUNK)                    // 16
#define BL        (BATCH*SEQLEN)                    // 8192
#define RMS_EPS   1e-5f

typedef __bf16 bf16_t;
typedef __attribute__((ext_vector_type(16))) __bf16 bf16x16;
typedef __attribute__((ext_vector_type(8)))  __bf16 bf16x8;
typedef __attribute__((ext_vector_type(8)))  float  f32x8;

// ---------------- small device math helpers ----------------
__device__ __forceinline__ float softplus_f(float x) {
    return (x > 20.f) ? x : log1pf(__expf(x));
}
__device__ __forceinline__ float silu_f(float x) {
    return x * __frcp_rn(1.f + __expf(-x));
}

// ---------------- WMMA fragment loaders (CDNA5 bf16 16x16x32 layouts) ----
// A: 16x32 (MxK).  lane<16 : row=lane,    K = {0..7, 16..23}
//                  lane>=16: row=lane-16, K = {8..15, 24..31}
__device__ __forceinline__ bf16x16 load_frag_a(const bf16_t* base, int ld, int lane) {
    int m  = lane & 15;
    int hi = (lane >> 4) * 8;
    const bf16_t* p = base + (long)m * ld + hi;
    union { bf16x16 v; bf16x8 h[2]; } u;
    u.h[0] = *(const bf16x8*)(p);
    u.h[1] = *(const bf16x8*)(p + 16);
    return u.v;
}
// B: 32x16 (KxN) fed from an N-major (B-transposed, K-contiguous) buffer.
__device__ __forceinline__ bf16x16 load_frag_b(const bf16_t* baseT, int ld, int lane) {
    int n  = lane & 15;
    int ks = (lane >> 4) * 8;
    const bf16_t* p = baseT + (long)n * ld + ks;
    union { bf16x16 v; bf16x8 h[2]; } u;
    u.h[0] = *(const bf16x8*)(p);
    u.h[1] = *(const bf16x8*)(p + 16);
    return u.v;
}
__device__ __forceinline__ f32x8 wmma_bf16(bf16x16 a, bf16x16 b, f32x8 c) {
    return __builtin_amdgcn_wmma_f32_16x16x32_bf16(false, a, false, b,
                                                   (short)0, c, false, false);
}

// async global->LDS copy of 16B per lane (VGLOBAL encoding, GV mode, ASYNCcnt)
__device__ __forceinline__ void async_copy_b128(const bf16_t* gsrc, bf16_t* ldst) {
    unsigned int       lds_addr = (unsigned int)(unsigned long long)(uintptr_t)ldst;
    unsigned long long gaddr    = (unsigned long long)(uintptr_t)gsrc;
    asm volatile("global_load_async_to_lds_b128 %0, %1, off"
                 :: "v"(lds_addr), "v"(gaddr) : "memory");
}
__device__ __forceinline__ void wait_asynccnt_le4() {
    asm volatile("s_wait_asynccnt 0x4" ::: "memory");
}
__device__ __forceinline__ void wait_asynccnt_0() {
    asm volatile("s_wait_asynccnt 0x0" ::: "memory");
}

// ======================================================================
// 1) f32 -> bf16 converters
// ======================================================================
__global__ __launch_bounds__(256) void cvt_bf16_kernel(const float* __restrict__ src,
                                                       bf16_t* __restrict__ dst, long n) {
    long i = (long)blockIdx.x * 256 + threadIdx.x;
    if (i < n) dst[i] = (bf16_t)src[i];
}

// W_in (4384x1024) -> padded (4480x1024), zero pad rows
__global__ __launch_bounds__(256) void cvt_win_pad_kernel(const float* __restrict__ W,
                                                          bf16_t* __restrict__ dst) {
    long i = (long)blockIdx.x * 256 + threadIdx.x;
    long n = (long)NPROJ_PAD * D_MODEL;
    if (i >= n) return;
    long r = i / D_MODEL;
    dst[i] = (r < D_IN_PROJ) ? (bf16_t)W[i] : (bf16_t)0.f;
}

// ======================================================================
// 2) Generic batched WMMA GEMM:  C[M,N](f32) = A[M,K](bf16) x Bt[N,K](bf16)^T
//    block = 256 thr = 8 waves (4M x 2N); wave tile = 32x64; block tile 128x128
//    Double-buffered async global->LDS pipeline (ASYNCcnt), WMMA from LDS.
// ======================================================================
#define GK 32   // K-step per LDS buffer
#define GLD 40  // padded LDS row stride (elements)

__global__ __launch_bounds__(256) void gemm_bf16_wmma_kernel(
        const bf16_t* __restrict__ A, const bf16_t* __restrict__ Bt,
        float* __restrict__ C, int M, int N, int K,
        long strideA, long strideB, long strideC) {
    __shared__ __align__(16) bf16_t As[2][128][GLD];
    __shared__ __align__(16) bf16_t Bs[2][128][GLD];

    A  += (long)blockIdx.z * strideA;
    Bt += (long)blockIdx.z * strideB;
    C  += (long)blockIdx.z * strideC;

    const int t    = threadIdx.x;
    const int lane = t & 31;
    const int wv   = t >> 5;
    const int wm   = wv & 3;            // 0..3
    const int wn   = wv >> 2;           // 0..1
    const int rowA0 = blockIdx.y * 128;
    const int colB0 = blockIdx.x * 128;
    const int m0l   = wm * 32;          // wave tile origin inside LDS block
    const int n0l   = wn * 64;

    // per-thread copy coordinates: 512 16B-chunks per (A,B) tile pair
    // chunk e (0..511): row = e>>2 (0..127), 16B segment = (e&3)*8 elements
    auto issue_tiles = [&](int buf, int kt) {
        const int kbase = kt * GK;
#pragma unroll
        for (int i = 0; i < 2; ++i) {
            int e  = t + i * 256;
            int r  = e >> 2;
            int cs = (e & 3) * 8;
            async_copy_b128(A  + (long)(rowA0 + r) * K + kbase + cs, &As[buf][r][cs]);
            async_copy_b128(Bt + (long)(colB0 + r) * K + kbase + cs, &Bs[buf][r][cs]);
        }
    };

    f32x8 acc[2][4];
#pragma unroll
    for (int i = 0; i < 2; ++i)
#pragma unroll
        for (int j = 0; j < 4; ++j) acc[i][j] = (f32x8)0.f;

    const int nkt = K / GK;
    issue_tiles(0, 0);
    for (int kt = 0; kt < nkt; ++kt) {
        const int buf = kt & 1;
        if (kt + 1 < nkt) {
            issue_tiles(buf ^ 1, kt + 1);
            wait_asynccnt_le4();        // current buffer's 4 async ops retired
        } else {
            wait_asynccnt_0();
        }
        __syncthreads();

        bf16x16 a0 = load_frag_a(&As[buf][m0l][0], GLD, lane);
        bf16x16 a1 = load_frag_a(&As[buf][m0l + 16][0], GLD, lane);
#pragma unroll
        for (int j = 0; j < 4; ++j) {
            bf16x16 bfr = load_frag_b(&Bs[buf][n0l + 16 * j][0], GLD, lane);
            acc[0][j] = wmma_bf16(a0, bfr, acc[0][j]);
            acc[1][j] = wmma_bf16(a1, bfr, acc[1][j]);
        }
        __syncthreads();                // waves done reading buf before next overwrite
    }

    // C/D layout: VGPR r -> row = 8*(lane>>4)+r ; col = lane&15
    const int cb = colB0 + n0l + (lane & 15);
    const int rb = rowA0 + m0l + 8 * (lane >> 4);
#pragma unroll
    for (int i = 0; i < 2; ++i)
#pragma unroll
        for (int j = 0; j < 4; ++j)
#pragma unroll
            for (int r = 0; r < 8; ++r)
                C[(long)(rb + i * 16 + r) * N + cb + 16 * j] = acc[i][j][r];
}

// ======================================================================
// 3) causal depthwise conv (width 4) + bias + SiLU, split into x / B / C
// ======================================================================
__global__ __launch_bounds__(256) void conv_silu_split_kernel(
        const float* __restrict__ zx, const float* __restrict__ convw,
        const float* __restrict__ convb,
        bf16_t* __restrict__ Xb, bf16_t* __restrict__ Bb, bf16_t* __restrict__ Cb) {
    long i = (long)blockIdx.x * 256 + threadIdx.x;
    if (i >= (long)BL * CONV_DIM) return;
    int c   = (int)(i % CONV_DIM);
    int row = (int)(i / CONV_DIM);
    int b   = row / SEQLEN;
    int l   = row % SEQLEN;
    float v = convb[c];
#pragma unroll
    for (int w = 0; w < D_CONV; ++w) {
        int lp = l - (D_CONV - 1) + w;
        if (lp >= 0)
            v += zx[(long)(b * SEQLEN + lp) * NPROJ_PAD + D_INNER + c] * convw[c * D_CONV + w];
    }
    v = silu_f(v);
    if (c < D_INNER)              Xb[(long)row * D_INNER + c] = (bf16_t)v;
    else if (c < D_INNER + D_STATE) Bb[(long)row * D_STATE + (c - D_INNER)] = (bf16_t)v;
    else                           Cb[(long)row * D_STATE + (c - D_INNER - D_STATE)] = (bf16_t)v;
}

// ======================================================================
// 4) dt = softplus(dt_raw + bias); cA = per-chunk cumsum(dt*A)
// ======================================================================
__global__ __launch_bounds__(256) void dt_cumsum_kernel(
        const float* __restrict__ zx, const float* __restrict__ dt_bias,
        const float* __restrict__ A_log,
        float* __restrict__ dtb, float* __restrict__ cAb) {
    int t = blockIdx.x * 256 + threadIdx.x;            // 0..1023
    if (t >= BATCH * NCHUNK * NHEADS) return;
    int h = t & (NHEADS - 1);
    int c = (t >> 5) & (NCHUNK - 1);
    int b = t >> 9;
    float Ah  = -__expf(A_log[h]);
    float bia = dt_bias[h];
    float run = 0.f;
    int row0 = b * SEQLEN + c * CHUNK;
    for (int l = 0; l < CHUNK; ++l) {
        int row = row0 + l;
        float dtv = softplus_f(zx[(long)row * NPROJ_PAD + D_INNER + CONV_DIM + h] + bia);
        run += dtv * Ah;
        dtb[(long)row * NHEADS + h] = dtv;
        cAb[(long)row * NHEADS + h] = run;
    }
}

// ======================================================================
// 5) Y_intra per (b,c,h): M[l,s] = CB[l,s]*exp(cA_l-cA_s)*dt_s (l>=s)
//    Y[l, h*64+p] = sum_s M[l,s] * X[s,p]  + D[h]*X[l,p]
// ======================================================================
__global__ __launch_bounds__(256) void y_intra_kernel(
        const float* __restrict__ CBm, const float* __restrict__ cAb,
        const float* __restrict__ dtb, const bf16_t* __restrict__ Xb,
        const float* __restrict__ Dv, float* __restrict__ Yf) {
    __shared__ __align__(16) bf16_t Ml[CHUNK][72];
    __shared__ __align__(16) bf16_t Xt[64][72];
    __shared__ float  cAs[64], dts[64];

    const int h = blockIdx.x, c = blockIdx.y, b = blockIdx.z;
    const int t = threadIdx.x, lane = t & 31, wv = t >> 5;
    const int row0 = b * SEQLEN + c * CHUNK;
    const float* cbrow = CBm + ((long)(b * NCHUNK + c)) * (CHUNK * CHUNK) + (long)t * CHUNK;
    const float cAl = cAb[(long)(row0 + t) * NHEADS + h];

    f32x8 acc[2][4];
#pragma unroll
    for (int i = 0; i < 2; ++i)
#pragma unroll
        for (int j = 0; j < 4; ++j) acc[i][j] = (f32x8)0.f;

    for (int sb = 0; sb < CHUNK; sb += 64) {
        if (t < 64) {
            cAs[t] = cAb[(long)(row0 + sb + t) * NHEADS + h];
            dts[t] = dtb[(long)(row0 + sb + t) * NHEADS + h];
        }
        __syncthreads();
#pragma unroll 4
        for (int s = 0; s < 64; ++s) {
            int sg = sb + s;
            float v = 0.f;
            if (t >= sg) v = cbrow[sg] * __expf(cAl - cAs[s]) * dts[s];
            Ml[t][s] = (bf16_t)v;
        }
#pragma unroll
        for (int i = 0; i < 16; ++i) {
            int e = t + i * 256;
            int s = e >> 6, p = e & 63;
            Xt[p][s] = Xb[(long)(row0 + sb + s) * D_INNER + h * HEADDIM + p];
        }
        __syncthreads();
#pragma unroll
        for (int k = 0; k < 64; k += 32) {
            bf16x16 a0 = load_frag_a(&Ml[wv * 32][k], 72, lane);
            bf16x16 a1 = load_frag_a(&Ml[wv * 32 + 16][k], 72, lane);
#pragma unroll
            for (int j = 0; j < 4; ++j) {
                bf16x16 bfr = load_frag_b(&Xt[16 * j][k], 72, lane);
                acc[0][j] = wmma_bf16(a0, bfr, acc[0][j]);
                acc[1][j] = wmma_bf16(a1, bfr, acc[1][j]);
            }
        }
        __syncthreads();
    }
    const float Dh = Dv[h];
#pragma unroll
    for (int i = 0; i < 2; ++i)
#pragma unroll
        for (int j = 0; j < 4; ++j) {
            int col = h * HEADDIM + 16 * j + (lane & 15);
            int rowb = row0 + wv * 32 + i * 16 + 8 * (lane >> 4);
#pragma unroll
            for (int r = 0; r < 8; ++r) {
                long idx = (long)(rowb + r) * D_INNER + col;
                Yf[idx] = acc[i][j][r] + Dh * (float)Xb[idx];
            }
        }
}

// ======================================================================
// 6) chunk states per (b,c,h): S[p,n] = sum_l (X[l,p]*dstate_l) * B[l,n]
// ======================================================================
__global__ __launch_bounds__(256) void states_kernel(
        const bf16_t* __restrict__ Xb, const bf16_t* __restrict__ Bb,
        const float* __restrict__ cAb, const float* __restrict__ dtb,
        float* __restrict__ St) {
    __shared__ __align__(16) bf16_t Xd[64][136];
    __shared__ __align__(16) bf16_t Btl[128][136];
    __shared__ float  dsb[128];

    const int h = blockIdx.x, c = blockIdx.y, b = blockIdx.z;
    const int t = threadIdx.x, lane = t & 31, wv = t >> 5;
    const int row0 = b * SEQLEN + c * CHUNK;
    const float cAlast = cAb[(long)(row0 + CHUNK - 1) * NHEADS + h];
    const int mt = wv >> 1;
    const int n0 = (wv & 1) * 64;

    f32x8 acc[4];
#pragma unroll
    for (int j = 0; j < 4; ++j) acc[j] = (f32x8)0.f;

    for (int lb = 0; lb < CHUNK; lb += 128) {
        if (t < 128) {
            int row = row0 + lb + t;
            dsb[t] = __expf(cAlast - cAb[(long)row * NHEADS + h]) * dtb[(long)row * NHEADS + h];
        }
        __syncthreads();
#pragma unroll
        for (int i = 0; i < 32; ++i) {
            int e = t + i * 256;
            int l = e >> 6, p = e & 63;
            float xv = (float)Xb[(long)(row0 + lb + l) * D_INNER + h * HEADDIM + p];
            Xd[p][l] = (bf16_t)(xv * dsb[l]);
        }
#pragma unroll
        for (int i = 0; i < 64; ++i) {
            int e = t + i * 256;
            int l = e >> 7, n = e & 127;
            Btl[n][l] = Bb[(long)(row0 + lb + l) * D_STATE + n];
        }
        __syncthreads();
#pragma unroll
        for (int k = 0; k < 128; k += 32) {
            bf16x16 a = load_frag_a(&Xd[mt * 16][k], 136, lane);
#pragma unroll
            for (int j = 0; j < 4; ++j) {
                bf16x16 bfr = load_frag_b(&Btl[n0 + 16 * j][k], 136, lane);
                acc[j] = wmma_bf16(a, bfr, acc[j]);
            }
        }
        __syncthreads();
    }
    long base = (((long)(b * NCHUNK + c) * NHEADS + h) * HEADDIM) * D_STATE;
    int prow = mt * 16 + 8 * (lane >> 4);
#pragma unroll
    for (int j = 0; j < 4; ++j) {
        int ncol = n0 + 16 * j + (lane & 15);
#pragma unroll
        for (int r = 0; r < 8; ++r)
            St[base + (long)(prow + r) * D_STATE + ncol] = acc[j][r];
    }
}

// ======================================================================
// 7) inter-chunk scan (16 steps, sequential)
// ======================================================================
__global__ __launch_bounds__(256) void chunk_scan_kernel(
        const float* __restrict__ St, const float* __restrict__ cAb,
        float* __restrict__ Pv) {
    int idx = blockIdx.x * 256 + threadIdx.x;
    if (idx >= BATCH * NHEADS * HEADDIM * D_STATE) return;
    int n = idx & 127, p = (idx >> 7) & 63, h = (idx >> 13) & 31, b = idx >> 18;
    float carry = 0.f;
    for (int c = 0; c < NCHUNK; ++c) {
        long sidx = (((long)(b * NCHUNK + c) * NHEADS + h) * HEADDIM + p) * D_STATE + n;
        Pv[sidx] = carry;
        int rl = b * SEQLEN + c * CHUNK + CHUNK - 1;
        float cd = __expf(cAb[(long)rl * NHEADS + h]);
        carry = carry * cd + St[sidx];
    }
}

// ======================================================================
// 8) Y_inter per (b,c,h): Y[l,h*64+p] += sum_n exp(cA_l)*C[l,n]*prev[p,n]
// ======================================================================
__global__ __launch_bounds__(256) void y_inter_kernel(
        const bf16_t* __restrict__ Cb, const float* __restrict__ cAb,
        const float* __restrict__ Pv, float* __restrict__ Yf) {
    __shared__ __align__(16) bf16_t Cs[128][136];
    __shared__ __align__(16) bf16_t Pt[64][136];
    __shared__ float  esc[128];

    const int h = blockIdx.x, c = blockIdx.y, b = blockIdx.z;
    const int t = threadIdx.x, lane = t & 31, wv = t >> 5;
    const int row0 = b * SEQLEN + c * CHUNK;
    const long pbase = (((long)(b * NCHUNK + c) * NHEADS + h) * HEADDIM) * D_STATE;

#pragma unroll
    for (int i = 0; i < 32; ++i) {
        int e = t + i * 256;
        int p = e >> 7, n = e & 127;
        Pt[p][n] = (bf16_t)Pv[pbase + (long)p * D_STATE + n];
    }
    for (int lb = 0; lb < CHUNK; lb += 128) {
        if (t < 128) esc[t] = __expf(cAb[(long)(row0 + lb + t) * NHEADS + h]);
        __syncthreads();
#pragma unroll
        for (int i = 0; i < 64; ++i) {
            int e = t + i * 256;
            int l = e >> 7, n = e & 127;
            Cs[l][n] = (bf16_t)((float)Cb[(long)(row0 + lb + l) * D_STATE + n] * esc[l]);
        }
        __syncthreads();
        f32x8 acc[4];
#pragma unroll
        for (int j = 0; j < 4; ++j) acc[j] = (f32x8)0.f;
#pragma unroll
        for (int k = 0; k < 128; k += 32) {
            bf16x16 a = load_frag_a(&Cs[wv * 16][k], 136, lane);
#pragma unroll
            for (int j = 0; j < 4; ++j) {
                bf16x16 bfr = load_frag_b(&Pt[16 * j][k], 136, lane);
                acc[j] = wmma_bf16(a, bfr, acc[j]);
            }
        }
        int rowb = row0 + lb + wv * 16 + 8 * (lane >> 4);
#pragma unroll
        for (int j = 0; j < 4; ++j) {
            int col = h * HEADDIM + 16 * j + (lane & 15);
#pragma unroll
            for (int r = 0; r < 8; ++r)
                Yf[(long)(rowb + r) * D_INNER + col] += acc[j][r];
        }
        __syncthreads();
    }
}

// ======================================================================
// 9) gate with silu(z), RMS norm over D_INNER, emit bf16 for out-proj
// ======================================================================
__global__ __launch_bounds__(256) void gate_rmsnorm_kernel(
        const float* __restrict__ Yf, const float* __restrict__ zx,
        const float* __restrict__ nw, bf16_t* __restrict__ Yb) {
    __shared__ float red[256];
    const int row = blockIdx.x, t = threadIdx.x;
    float vals[8];
    float ss = 0.f;
#pragma unroll
    for (int i = 0; i < 8; ++i) {
        int ccol = t + i * 256;
        float zv = zx[(long)row * NPROJ_PAD + ccol];
        float yv = Yf[(long)row * D_INNER + ccol] * silu_f(zv);
        vals[i] = yv;
        ss += yv * yv;
    }
    red[t] = ss;
    __syncthreads();
    for (int s = 128; s > 0; s >>= 1) {
        if (t < s) red[t] += red[t + s];
        __syncthreads();
    }
    float rms = rsqrtf(red[0] / (float)D_INNER + RMS_EPS);
#pragma unroll
    for (int i = 0; i < 8; ++i) {
        int ccol = t + i * 256;
        Yb[(long)row * D_INNER + ccol] = (bf16_t)(vals[i] * rms * nw[ccol]);
    }
}

// ======================================================================
// host-side launch
// ======================================================================
extern "C" void kernel_launch(void* const* d_in, const int* in_sizes, int n_in,
                              void* d_out, int out_size, void* d_ws, size_t ws_size,
                              hipStream_t stream) {
    const float* u       = (const float*)d_in[0];
    const float* W_in    = (const float*)d_in[1];
    const float* conv_w  = (const float*)d_in[2];
    const float* conv_b  = (const float*)d_in[3];
    const float* dt_bias = (const float*)d_in[4];
    const float* A_log   = (const float*)d_in[5];
    const float* Dv      = (const float*)d_in[6];
    const float* norm_w  = (const float*)d_in[7];
    const float* W_out   = (const float*)d_in[8];
    float* out = (float*)d_out;

    char* ws = (char*)d_ws;
    size_t off = 0;
    auto carve = [&](size_t bytes) {
        char* p = ws + off;
        off = (off + bytes + 255) & ~(size_t)255;
        return p;
    };
    bf16_t* UB  = (bf16_t*)carve((size_t)BL * D_MODEL * 2);
    bf16_t* WB  = (bf16_t*)carve((size_t)NPROJ_PAD * D_MODEL * 2);
    bf16_t* WOB = (bf16_t*)carve((size_t)D_MODEL * D_INNER * 2);
    float*  ZX  = (float*)carve((size_t)BL * NPROJ_PAD * 4);
    bf16_t* XB  = (bf16_t*)carve((size_t)BL * D_INNER * 2);
    bf16_t* BB  = (bf16_t*)carve((size_t)BL * D_STATE * 2);
    bf16_t* CB_ = (bf16_t*)carve((size_t)BL * D_STATE * 2);
    float*  DT  = (float*)carve((size_t)BL * NHEADS * 4);
    float*  CA  = (float*)carve((size_t)BL * NHEADS * 4);
    float*  CBM = (float*)carve((size_t)BATCH * NCHUNK * CHUNK * CHUNK * 4);
    float*  ST  = (float*)carve((size_t)BATCH * NCHUNK * NHEADS * HEADDIM * D_STATE * 4);
    float*  PV  = (float*)carve((size_t)BATCH * NCHUNK * NHEADS * HEADDIM * D_STATE * 4);
    float*  YF  = (float*)carve((size_t)BL * D_INNER * 4);
    bf16_t* YB  = (bf16_t*)carve((size_t)BL * D_INNER * 2);

    {
        long n = (long)BL * D_MODEL;
        cvt_bf16_kernel<<<dim3((n + 255) / 256), dim3(256), 0, stream>>>(u, UB, n);
    }
    {
        long n = (long)NPROJ_PAD * D_MODEL;
        cvt_win_pad_kernel<<<dim3((n + 255) / 256), dim3(256), 0, stream>>>(W_in, WB);
    }
    {
        long n = (long)D_MODEL * D_INNER;
        cvt_bf16_kernel<<<dim3((n + 255) / 256), dim3(256), 0, stream>>>(W_out, WOB, n);
    }
    // in_proj: ZX[8192,4480] = UB[8192,1024] x WB[4480,1024]^T
    gemm_bf16_wmma_kernel<<<dim3(NPROJ_PAD / 128, BL / 128, 1), dim3(256), 0, stream>>>(
        UB, WB, ZX, BL, NPROJ_PAD, D_MODEL, 0, 0, 0);
    {
        long n = (long)BL * CONV_DIM;
        conv_silu_split_kernel<<<dim3((n + 255) / 256), dim3(256), 0, stream>>>(
            ZX, conv_w, conv_b, XB, BB, CB_);
    }
    dt_cumsum_kernel<<<dim3((BATCH * NCHUNK * NHEADS + 255) / 256), dim3(256), 0, stream>>>(
        ZX, dt_bias, A_log, DT, CA);
    // CB = C x B^T per (b,chunk)
    gemm_bf16_wmma_kernel<<<dim3(CHUNK / 128, CHUNK / 128, BATCH * NCHUNK), dim3(256), 0, stream>>>(
        CB_, BB, CBM, CHUNK, CHUNK, D_STATE,
        (long)CHUNK * D_STATE, (long)CHUNK * D_STATE, (long)CHUNK * CHUNK);
    y_intra_kernel<<<dim3(NHEADS, NCHUNK, BATCH), dim3(256), 0, stream>>>(
        CBM, CA, DT, XB, Dv, YF);
    states_kernel<<<dim3(NHEADS, NCHUNK, BATCH), dim3(256), 0, stream>>>(
        XB, BB, CA, DT, ST);
    chunk_scan_kernel<<<dim3((BATCH * NHEADS * HEADDIM * D_STATE + 255) / 256), dim3(256), 0, stream>>>(
        ST, CA, PV);
    y_inter_kernel<<<dim3(NHEADS, NCHUNK, BATCH), dim3(256), 0, stream>>>(
        CB_, CA, PV, YF);
    gate_rmsnorm_kernel<<<dim3(BL), dim3(256), 0, stream>>>(YF, ZX, norm_w, YB);
    // out_proj: out[8192,1024] = YB[8192,2048] x WOB[1024,2048]^T
    gemm_bf16_wmma_kernel<<<dim3(D_MODEL / 128, BL / 128, 1), dim3(256), 0, stream>>>(
        YB, WOB, out, BL, D_MODEL, D_INNER, 0, 0, 0);
}
